// NTXentLoss_20804821582531
// MI455X (gfx1250) — compile-verified, matrix operand order checked by reading, not run on hardware
//
#include <hip/hip_runtime.h>
#include <hip/hip_bf16.h>
#include <math.h>

// ---------- types for CDNA5 WMMA ----------
typedef __bf16 v16bf __attribute__((ext_vector_type(16)));
typedef __bf16 v8bf  __attribute__((ext_vector_type(8)));
typedef float  v8f   __attribute__((ext_vector_type(8)));

#define N_ROWS 4096
#define D_K    1024
#define TEMP_INV 2.0f           // 1/T, T = 0.5
#define PAIR_COUNT 16773120.0f  // 4096*4095 (both triangles concatenated)

// Block tiling: 8 waves = 2 row-waves x 4 col-waves; each wave owns a 64x32
// patch as 4x2 WMMA tiles -> block tile 128 x 128.
#define BLK_ROWS 128
#define BLK_COLS 128

// ---------- helpers ----------
static __device__ __forceinline__ unsigned short f32_to_bf16_rne(float f) {
    union { float f; unsigned u; } v; v.f = f;
    unsigned r = v.u + 0x7FFFu + ((v.u >> 16) & 1u);
    return (unsigned short)(r >> 16);
}

// Load one 16x32 bf16 WMMA operand fragment from a row-major [N_ROWS][D_K]
// matrix. Per ISA 7.12.2 (16-bit A-matrix 16x32):
//   lanes 0-15 : row = row0+lane,    halves [0..7]=K0..K0+7,   [8..15]=K0+16..23
//   lanes 16-31: row = row0+lane-16, halves [0..7]=K0+8..15,   [8..15]=K0+24..31
// The B operand of C = A * B^T has the same per-lane addressing (lane = column,
// which is a row of the row-major source), so this loader serves both.
static __device__ __forceinline__ v16bf load_frag(const __bf16* __restrict__ base,
                                                  int row0, int k0, int lane) {
    int r = row0 + (lane & 15);
    const __bf16* p = base + (size_t)r * D_K + k0 + ((lane >> 4) << 3);
    v8bf lo = *reinterpret_cast<const v8bf*>(p);        // 16B -> global_load_b128
    v8bf hi = *reinterpret_cast<const v8bf*>(p + 16);
    return __builtin_shufflevector(lo, hi, 0,1,2,3,4,5,6,7,8,9,10,11,12,13,14,15);
}

static __device__ __forceinline__ v8f wmma_bf16(v16bf a, v16bf b, v8f c) {
    return __builtin_amdgcn_wmma_f32_16x16x32_bf16(
        /*neg_a=*/false, a, /*neg_b=*/false, b,
        /*c_mod=*/(short)0, c, /*reuse_a=*/false, /*reuse_b=*/false);
}

static __device__ __forceinline__ void block_reduce_add(float local, float* gout) {
    __shared__ float red[256];
    int tid = threadIdx.x;
    red[tid] = local;
    __syncthreads();
    for (int s = 128; s > 0; s >>= 1) {
        if (tid < s) red[tid] += red[tid + s];
        __syncthreads();
    }
    if (tid == 0) atomicAdd(gout, red[0]);
}

// ---------- kernels ----------
__global__ void init_acc_kernel(float* acc) {
    acc[0] = 0.0f;  // sum exp(cross/T - 2)
    acc[1] = 0.0f;  // sum of log1p terms
    acc[2] = 0.0f;  // lse_cross
}

// One block (256 threads) per row: L2-normalize and cast to bf16.
__global__ void normalize_kernel(const float* __restrict__ x,
                                 unsigned short* __restrict__ out) {
    __shared__ float red[256];
    int row = blockIdx.x;
    int tid = threadIdx.x;
    const float* rp = x + (size_t)row * D_K;
    float v0 = rp[tid], v1 = rp[tid + 256], v2 = rp[tid + 512], v3 = rp[tid + 768];
    red[tid] = v0 * v0 + v1 * v1 + v2 * v2 + v3 * v3;
    __syncthreads();
    for (int s = 128; s > 0; s >>= 1) {
        if (tid < s) red[tid] += red[tid + s];
        __syncthreads();
    }
    float inv = 1.0f / fmaxf(sqrtf(red[0]), 1e-8f);
    unsigned short* op = out + (size_t)row * D_K;
    op[tid]       = f32_to_bf16_rne(v0 * inv);
    op[tid + 256] = f32_to_bf16_rne(v1 * inv);
    op[tid + 512] = f32_to_bf16_rne(v2 * inv);
    op[tid + 768] = f32_to_bf16_rne(v3 * inv);
}

// Shared 64x32-per-wave WMMA macro-tile: 6 fragment loads -> 8 WMMAs per
// K-step. ~130 VGPRs live (8 accums + 6 frags) -> no spills.
static __device__ __forceinline__ void gemm_64x32(const __bf16* __restrict__ A,
                                                  const __bf16* __restrict__ B,
                                                  int rowBase, int colBase, int lane,
                                                  v8f acc[4][2]) {
#pragma unroll 1
    for (int k0 = 0; k0 < D_K; k0 += 32) {
        v16bf af[4], bf[2];
#pragma unroll
        for (int r = 0; r < 4; ++r) af[r] = load_frag(A, rowBase + r * 16, k0, lane);
#pragma unroll
        for (int c = 0; c < 2; ++c) bf[c] = load_frag(B, colBase + c * 16, k0, lane);
#pragma unroll
        for (int r = 0; r < 4; ++r)
#pragma unroll
            for (int c = 0; c < 2; ++c)
                acc[r][c] = wmma_bf16(af[r], bf[c], acc[r][c]);
    }
}

// cross = a_norm @ s_norm^T ; accumulate sum(exp(sim/T - 2)) globally.
__global__ void __launch_bounds__(256, 1)
cross_sum_kernel(const __bf16* __restrict__ A,
                 const __bf16* __restrict__ S,
                 float* __restrict__ acc) {
    int lane = threadIdx.x & 31;
    int wid  = threadIdx.x >> 5;
    int rowBase = blockIdx.y * BLK_ROWS + (wid >> 2) * 64;
    int colBase = blockIdx.x * BLK_COLS + (wid & 3) * 32;

    v8f c[4][2] = {};
    gemm_64x32(A, S, rowBase, colBase, lane, c);

    float local = 0.0f;
#pragma unroll
    for (int r = 0; r < 4; ++r)
#pragma unroll
        for (int cc = 0; cc < 2; ++cc)
#pragma unroll
            for (int e = 0; e < 8; ++e)
                local += __expf(c[r][cc][e] * TEMP_INV - 2.0f);
    block_reduce_add(local, &acc[0]);
}

__global__ void finalize_lse_kernel(float* acc) {
    acc[2] = logf(acc[0]) + 2.0f;   // lse_cross = log(sum exp(x-2)) + 2
}

// Gram = x_norm @ x_norm^T ; for each strict upper-triangle element add
// log1p(exp(L - sim/T)).
__global__ void __launch_bounds__(256, 1)
gram_loss_kernel(const __bf16* __restrict__ X,
                 const float* __restrict__ acc,
                 float* __restrict__ lossAcc) {
    int blockRow = blockIdx.y * BLK_ROWS;            // i in [blockRow, +127]
    int blockCol = blockIdx.x * BLK_COLS;            // j in [blockCol, +127]
    if (blockCol + BLK_COLS - 1 <= blockRow) return; // uniform: no j > i here

    int lane = threadIdx.x & 31;
    int wid  = threadIdx.x >> 5;
    int rowBase = blockRow + (wid >> 2) * 64;
    int colBase = blockCol + (wid & 3) * 32;

    v8f c[4][2] = {};
    gemm_64x32(X, X, rowBase, colBase, lane, c);

    // C/D layout (ISA 7.12.2): lane 0-15 -> M = vgpr, lane 16-31 -> M = vgpr+8;
    // N = lane % 16.
    float L = acc[2];
    int mOff = (lane >> 4) << 3;   // 0 or 8
    int nOff = lane & 15;
    float local = 0.0f;
#pragma unroll
    for (int r = 0; r < 4; ++r)
#pragma unroll
        for (int cc = 0; cc < 2; ++cc) {
            int i0 = rowBase + r * 16 + mOff;
            int j0 = colBase + cc * 16 + nOff;
#pragma unroll
            for (int e = 0; e < 8; ++e)
                if (j0 > i0 + e)
                    local += log1pf(__expf(L - c[r][cc][e] * TEMP_INV));
        }
    block_reduce_add(local, lossAcc);
}

__global__ void final_mean_kernel(const float* acc, float* out) {
    out[0] = acc[1] / PAIR_COUNT;
}

// ---------- launch ----------
extern "C" void kernel_launch(void* const* d_in, const int* in_sizes, int n_in,
                              void* d_out, int out_size, void* d_ws, size_t ws_size,
                              hipStream_t stream) {
    const float* stereos  = (const float*)d_in[0];
    const float* astereos = (const float*)d_in[1];
    float* out = (float*)d_out;

    // workspace layout: [0..255] accumulators, then two bf16 matrices.
    float* acc = (float*)d_ws;
    unsigned short* sN = (unsigned short*)((char*)d_ws + 256);
    unsigned short* aN = sN + (size_t)N_ROWS * D_K;

    init_acc_kernel<<<1, 1, 0, stream>>>(acc);
    normalize_kernel<<<N_ROWS, 256, 0, stream>>>(stereos,  sN);
    normalize_kernel<<<N_ROWS, 256, 0, stream>>>(astereos, aN);

    dim3 grid(N_ROWS / BLK_COLS, N_ROWS / BLK_ROWS);   // 32 x 32 blocks
    cross_sum_kernel<<<grid, 256, 0, stream>>>((const __bf16*)aN, (const __bf16*)sN, acc);
    finalize_lse_kernel<<<1, 1, 0, stream>>>(acc);
    gram_loss_kernel<<<grid, 256, 0, stream>>>((const __bf16*)sN, acc, &acc[1]);
    gram_loss_kernel<<<grid, 256, 0, stream>>>((const __bf16*)aN, acc, &acc[1]);
    final_mean_kernel<<<1, 1, 0, stream>>>(acc, out);
}
